// MCL_MSG3D_69372311765178
// MI455X (gfx1250) — matrix operand order repaired and checked by reading, not compiled, not run on hardware
//
#include <hip/hip_runtime.h>
#include <hip/hip_bf16.h>

// ---------------------------------------------------------------------------
// Problem constants (from reference): B=32, C=H=384, T=75, V=25, K=3
// ---------------------------------------------------------------------------
#define BB   32
#define CC   384
#define TT   75
#define VV   25
#define HH   384
#define KK   3
#define NOUT 4608          // 4*H*K
#define KIN  2304          // 2*C*3 (im2col K for the fused [x,h] conv)
#define MROWS (BB*VV)      // 800 GEMM rows per timestep
#define NELEM (CC*TT*VV)   // 720000 elements per sample per view

typedef __bf16 v16bf __attribute__((ext_vector_type(16)));
typedef float  v8f   __attribute__((ext_vector_type(8)));

__device__ __forceinline__ unsigned short f2bf(float f) {
    unsigned int u = __builtin_bit_cast(unsigned int, f);
    u += 0x7FFFu + ((u >> 16) & 1u);   // round-to-nearest-even
    return (unsigned short)(u >> 16);
}

// ---------------------------------------------------------------------------
// mask factors: mf[j][v][c] = tanh(mask_j[v,c]) + 1
// ---------------------------------------------------------------------------
__global__ void mprep_kernel(const float* __restrict__ m1, const float* __restrict__ m2,
                             const float* __restrict__ m3, float* __restrict__ mf) {
    int idx = blockIdx.x * blockDim.x + threadIdx.x;
    if (idx >= 3 * VV * CC) return;
    int j = idx / (VV * CC);
    int r = idx - j * (VV * CC);
    const float* src = (j == 0) ? m1 : (j == 1) ? m2 : m3;
    mf[idx] = tanhf(src[r]) + 1.0f;
}

// ---------------------------------------------------------------------------
// Weight repack, TRANSPOSED for the B-operand: WcT[n][k] (bf16),
// k = tap*768 + cin, from conv_w (4608, 768, 3).  K-contiguous rows let a
// WMMA B fragment be two 16-byte LDS reads per lane.
// ---------------------------------------------------------------------------
__global__ void wprep_kernel(const float* __restrict__ w, unsigned short* __restrict__ WcT) {
    int idx = blockIdx.x * blockDim.x + threadIdx.x;
    if (idx >= NOUT * KIN) return;
    int n = idx / KIN;
    int k = idx - n * KIN;
    int tap = k / (2 * CC);
    int cin = k - tap * (2 * CC);
    WcT[idx] = f2bf(w[(n * (2 * CC) + cin) * KK + tap]);
}

// ---------------------------------------------------------------------------
// Zero-init LSTM state (c fp32, h bf16) — ws is poisoned, must clear each call
// ---------------------------------------------------------------------------
__global__ void init_state_kernel(float* __restrict__ c, unsigned short* __restrict__ hbf) {
    int idx = blockIdx.x * blockDim.x + threadIdx.x;
    if (idx < BB * HH * VV) { c[idx] = 0.0f; hbf[idx] = 0; }
}

// ---------------------------------------------------------------------------
// LayerNorm over (C,T,V) per sample for 3 views + mask + elementwise max.
// Output bf16, layout (B, T, V, C) so conv patches are contiguous channels.
// One block per batch sample.
// ---------------------------------------------------------------------------
__global__ __launch_bounds__(256)
void ln_mask_max_kernel(const float* __restrict__ x1, const float* __restrict__ x2,
                        const float* __restrict__ x3, const float* __restrict__ mf,
                        unsigned short* __restrict__ xnorm) {
    const int b = blockIdx.x;
    const int tid = threadIdx.x;
    const float* xs0 = x1 + (size_t)b * NELEM;
    const float* xs1 = x2 + (size_t)b * NELEM;
    const float* xs2 = x3 + (size_t)b * NELEM;

    double s0 = 0, s1 = 0, s2 = 0, q0 = 0, q1 = 0, q2 = 0;
    for (int i = tid; i < NELEM; i += 256) {
        float a = xs0[i], bq = xs1[i], cq = xs2[i];
        s0 += a;  q0 += (double)a * a;
        s1 += bq; q1 += (double)bq * bq;
        s2 += cq; q2 += (double)cq * cq;
    }
    __shared__ double red[256 * 6];
    red[tid * 6 + 0] = s0; red[tid * 6 + 1] = s1; red[tid * 6 + 2] = s2;
    red[tid * 6 + 3] = q0; red[tid * 6 + 4] = q1; red[tid * 6 + 5] = q2;
    __syncthreads();
    for (int off = 128; off > 0; off >>= 1) {
        if (tid < off)
            for (int j = 0; j < 6; ++j)
                red[tid * 6 + j] += red[(tid + off) * 6 + j];
        __syncthreads();
    }
    __shared__ float mu[3], rs[3];
    if (tid == 0) {
        for (int j = 0; j < 3; ++j) {
            double mean = red[j] / (double)NELEM;
            double var  = red[3 + j] / (double)NELEM - mean * mean;
            mu[j] = (float)mean;
            rs[j] = (float)(1.0 / sqrt(var + 1e-5));
        }
    }
    __syncthreads();

    const float mu0 = mu[0], mu1 = mu[1], mu2 = mu[2];
    const float r0 = rs[0], r1 = rs[1], r2 = rs[2];
    for (int i = tid; i < NELEM; i += 256) {
        int c  = i / (TT * VV);
        int r  = i - c * (TT * VV);
        int t  = r / VV;
        int v  = r - t * VV;
        int mi = v * CC + c;
        float a0 = (xs0[i] - mu0) * r0 * mf[mi];
        float a1 = (xs1[i] - mu1) * r1 * mf[VV * CC + mi];
        float a2 = (xs2[i] - mu2) * r2 * mf[2 * VV * CC + mi];
        float best = fmaxf(a0, fmaxf(a1, a2));
        xnorm[((size_t)(b * TT + t) * VV + v) * CC + c] = f2bf(best);
    }
}

// ---------------------------------------------------------------------------
// Per-timestep implicit-GEMM (im2col over V, k=3, pad=1, fused [x_t, h]):
//   y[row=(b*25+v)][n] = bias[n] + sum_k patch(row,k) * W[k][n]
// M=800, N=4608, K=2304.  bf16 WMMA, f32 accumulate.
//
// Block: 256 threads = 8 waves; block tile 32(M) x 256(N), K-chunk 64.
// Each wave computes a 32x32 patch: per chunk 2 k-halves x (2 A x 2 B frags)
// = 8 WMMA between one barrier pair.  LDS layouts are K-contiguous so
// fragment loads are pure ds_load_b128:
//   A frag (lane=row M):   two 16B runs  k[8h..8h+8), k[16+8h..16+8h+8)
//   B frag (lane=col N):   two 16B runs  k[8h..8h+8), k[16+8h..16+8h+8)   (weights transposed)
// ---------------------------------------------------------------------------
#define MTILE  32
#define NTILE  256
#define KCH    64
#define LDK    72     // padded K-row stride (ushorts): 144B = 9x16B, bank-spread

__global__ __launch_bounds__(256)
void step_gemm_kernel(const unsigned short* __restrict__ xnorm,
                      const unsigned short* __restrict__ hbf,
                      const unsigned short* __restrict__ WcT,
                      const float* __restrict__ bias,
                      float* __restrict__ y, int t) {
    __shared__ unsigned short lA[MTILE * LDK];     // 32 rows(M)   x 64 k (padded)
    __shared__ unsigned short lB[NTILE * LDK];     // 256 rows(N)  x 64 k (padded)

    const int tid   = threadIdx.x;
    const int wave  = tid >> 5;
    const int lane  = tid & 31;
    const int l16   = lane & 15;
    const int hf    = lane >> 4;
    const int mBase = blockIdx.x * MTILE;
    const int nBase = blockIdx.y * NTILE;

    v8f acc00 = {0,0,0,0,0,0,0,0}, acc01 = acc00, acc10 = acc00, acc11 = acc00;

    for (int kc = 0; kc < KIN; kc += KCH) {
        if (kc + KCH < KIN)
            __builtin_prefetch(WcT + (size_t)(nBase + wave * 32) * KIN + kc + KCH, 0, 1);

        // --- stage A patch: 32 rows x 64 k (gathered im2col over x/h) ---
        for (int i = tid; i < MTILE * KCH; i += 256) {
            int m  = i >> 6;
            int kk = i & 63;
            int k  = kc + kk;
            int tap = k / (2 * CC);
            int cin = k - tap * (2 * CC);
            int r = mBase + m;
            int b = r / VV;
            int v = r - b * VV;
            int vv = v + tap - 1;
            unsigned short val = 0;
            if (vv >= 0 && vv < VV) {
                val = (cin < CC)
                    ? xnorm[((size_t)(b * TT + t) * VV + vv) * CC + cin]
                    : hbf[((size_t)b * VV + vv) * CC + (cin - CC)];
            }
            lA[m * LDK + kk] = val;
        }
        // --- stage B tile: 256 N-rows x 64 k, straight 16B vector copies ---
        for (int i = tid; i < NTILE * 8; i += 256) {
            int row = i >> 3;
            int seg = i & 7;
            uint4 d = *(const uint4*)(WcT + (size_t)(nBase + row) * KIN + kc + seg * 8);
            *(uint4*)(lB + row * LDK + seg * 8) = d;
        }
        __syncthreads();

        // --- 2 k-halves x 4 WMMA, fragments as pure b128 LDS loads ---
        union Frag { uint4 q[2]; v16bf v; };
#pragma unroll
        for (int kh = 0; kh < 2; ++kh) {
            const int ko = kh * 32;
            Frag fa0, fa1, fb0, fb1;
            const unsigned short* pa0 = lA + (l16) * LDK + ko;
            const unsigned short* pa1 = lA + (16 + l16) * LDK + ko;
            fa0.q[0] = *(const uint4*)(pa0 + 8 * hf);
            fa0.q[1] = *(const uint4*)(pa0 + 16 + 8 * hf);
            fa1.q[0] = *(const uint4*)(pa1 + 8 * hf);
            fa1.q[1] = *(const uint4*)(pa1 + 16 + 8 * hf);
            const unsigned short* pb0 = lB + (wave * 32 + l16) * LDK + ko;
            const unsigned short* pb1 = lB + (wave * 32 + 16 + l16) * LDK + ko;
            fb0.q[0] = *(const uint4*)(pb0 + 8 * hf);
            fb0.q[1] = *(const uint4*)(pb0 + 16 + 8 * hf);
            fb1.q[0] = *(const uint4*)(pb1 + 8 * hf);
            fb1.q[1] = *(const uint4*)(pb1 + 16 + 8 * hf);

            acc00 = __builtin_amdgcn_wmma_f32_16x16x32_bf16(false, fa0.v, false, fb0.v, (short)0, acc00, false, false);
            acc01 = __builtin_amdgcn_wmma_f32_16x16x32_bf16(false, fa0.v, false, fb1.v, (short)0, acc01, false, false);
            acc10 = __builtin_amdgcn_wmma_f32_16x16x32_bf16(false, fa1.v, false, fb0.v, (short)0, acc10, false, false);
            acc11 = __builtin_amdgcn_wmma_f32_16x16x32_bf16(false, fa1.v, false, fb1.v, (short)0, acc11, false, false);
        }
        __syncthreads();
    }

    // --- epilogue: + bias, store y[row][n] ---
    const int col0 = nBase + wave * 32 + l16;
    const int col1 = col0 + 16;
    const float bv0 = bias[col0];
    const float bv1 = bias[col1];
#pragma unroll
    for (int j = 0; j < 8; ++j) {
        int row0 = mBase + j + 8 * hf;        // C/D layout: VGPR j -> M=j (+8 hi half)
        int row1 = row0 + 16;
        y[(size_t)row0 * NOUT + col0] = acc00[j] + bv0;
        y[(size_t)row0 * NOUT + col1] = acc01[j] + bv1;
        y[(size_t)row1 * NOUT + col0] = acc10[j] + bv0;
        y[(size_t)row1 * NOUT + col1] = acc11[j] + bv1;
    }
}

// ---------------------------------------------------------------------------
// Gate kernel: z[b,c,w] = sum_{k,v} y[b,v,k*1536+c] * A[k,v,w]; LSTM update.
// One thread per (b, hc, w) computes all 4 gates.
// ---------------------------------------------------------------------------
__global__ __launch_bounds__(256)
void gate_kernel(const float* __restrict__ y, const float* __restrict__ A,
                 float* __restrict__ c, unsigned short* __restrict__ hbf,
                 float* __restrict__ out, int t) {
    __shared__ float sA[KK * VV * VV];          // 1875 floats
    for (int i = threadIdx.x; i < KK * VV * VV; i += blockDim.x) sA[i] = A[i];
    __syncthreads();

    int idx = blockIdx.x * blockDim.x + threadIdx.x;
    if (idx >= BB * HH * VV) return;
    int w   = idx % VV;
    int rem = idx / VV;
    int hc  = rem % HH;
    int b   = rem / HH;

    const float* yb = y + (size_t)b * VV * NOUT;
    float z0 = 0.f, z1 = 0.f, z2 = 0.f, z3 = 0.f;
    for (int k = 0; k < KK; ++k) {
        const int nb = k * 4 * HH + hc;
        for (int v = 0; v < VV; ++v) {
            float a = sA[(k * VV + v) * VV + w];
            const float* yr = yb + (size_t)v * NOUT + nb;
            z0 = fmaf(yr[0],        a, z0);
            z1 = fmaf(yr[HH],       a, z1);
            z2 = fmaf(yr[2 * HH],   a, z2);
            z3 = fmaf(yr[3 * HH],   a, z3);
        }
    }
    float ig = 1.f / (1.f + __expf(-z0));
    float fg = 1.f / (1.f + __expf(-z1));
    float og = 1.f / (1.f + __expf(-z2));
    float gg = tanhf(z3);
    float cn = fg * c[idx] + ig * gg;
    float hn = og * tanhf(cn);
    c[idx] = cn;
    hbf[((size_t)b * VV + w) * HH + hc] = f2bf(hn);
    out[(((size_t)b * HH + hc) * TT + t) * VV + w] = hn;
}

// ---------------------------------------------------------------------------
// Launch
// ---------------------------------------------------------------------------
extern "C" void kernel_launch(void* const* d_in, const int* in_sizes, int n_in,
                              void* d_out, int out_size, void* d_ws, size_t ws_size,
                              hipStream_t stream) {
    const float* x1     = (const float*)d_in[0];
    const float* x2     = (const float*)d_in[1];
    const float* x3     = (const float*)d_in[2];
    const float* A      = (const float*)d_in[3];
    const float* conv_w = (const float*)d_in[4];
    const float* conv_b = (const float*)d_in[5];
    const float* mask1  = (const float*)d_in[6];
    const float* mask2  = (const float*)d_in[7];
    const float* mask3  = (const float*)d_in[8];
    float* out = (float*)d_out;

    // workspace carve-up (256B aligned)
    char* ws = (char*)d_ws;
    size_t off = 0;
    auto carve = [&](size_t bytes) { char* p = ws + off; off = (off + bytes + 255) & ~(size_t)255; return p; };
    float*          mf    = (float*)         carve(3 * VV * CC * sizeof(float));
    unsigned short* xnorm = (unsigned short*)carve((size_t)BB * TT * VV * CC * 2);
    unsigned short* WcT   = (unsigned short*)carve((size_t)NOUT * KIN * 2);
    unsigned short* hbf   = (unsigned short*)carve((size_t)BB * VV * HH * 2);
    float*          cbuf  = (float*)         carve((size_t)BB * HH * VV * sizeof(float));
    float*          ybuf  = (float*)         carve((size_t)BB * VV * NOUT * sizeof(float));
    (void)ws_size; (void)n_in; (void)in_sizes; (void)out_size;

    // preprocessing
    mprep_kernel<<<(3 * VV * CC + 255) / 256, 256, 0, stream>>>(mask1, mask2, mask3, mf);
    wprep_kernel<<<(NOUT * KIN + 255) / 256, 256, 0, stream>>>(conv_w, WcT);
    init_state_kernel<<<(BB * HH * VV + 255) / 256, 256, 0, stream>>>(cbuf, hbf);
    ln_mask_max_kernel<<<BB, 256, 0, stream>>>(x1, x2, x3, mf, xnorm);

    // 75-step recurrence
    dim3 ggrid(MROWS / MTILE, NOUT / NTILE);   // (25, 18)
    for (int t = 0; t < TT; ++t) {
        step_gemm_kernel<<<ggrid, 256, 0, stream>>>(xnorm, hbf, WcT, conv_b, ybuf, t);
        gate_kernel<<<(BB * HH * VV + 255) / 256, 256, 0, stream>>>(ybuf, A, cbuf, hbf, out, t);
    }
}
